// CRF_56873956933772
// MI455X (gfx1250) — matrix operand order repaired and checked by reading, not compiled, not run on hardware
//
#include <hip/hip_runtime.h>
#include <cmath>

#define B_ 1024
#define T_ 512
#define K_ 64
#define UP 72   // LDS U-buffer row pitch in halves (pad to dodge bank conflicts)

typedef _Float16 f16;
typedef __attribute__((ext_vector_type(16))) _Float16 v16h;
typedef __attribute__((ext_vector_type(8)))  _Float16 v8h;
typedef __attribute__((ext_vector_type(2)))  _Float16 v2h;
typedef __attribute__((ext_vector_type(8)))  float    v8f;

static constexpr size_t XBYTES  = (size_t)B_ * T_ * K_ * sizeof(f16); // 64 MB
static constexpr size_t MEBYTES = (size_t)B_ * T_ * sizeof(float);    // 2 MB

// ---------------------------------------------------------------------------
// Kernel 1: per (b,t) row: ME = max_j em ; X = exp(em - ME) in f16, stored
// tiled as X[tile][t][bi][j] so the scan reads contiguous 16B per lane.
// One wave (32 lanes) per row, 2 elements per lane.
// ---------------------------------------------------------------------------
__global__ void crf_prep(const float* __restrict__ em, f16* __restrict__ X,
                         float* __restrict__ ME) {
  int w = threadIdx.x >> 5, lane = threadIdx.x & 31;
  int row = blockIdx.x * 8 + w;              // [0, B*T)
  int b = row >> 9, t = row & (T_ - 1);
  const float* e = em + ((size_t)b * T_ + t) * K_ + 2 * lane;
  float v0 = e[0], v1 = e[1];
  float m = fmaxf(v0, v1);
  for (int d = 1; d < 32; d <<= 1) m = fmaxf(m, __shfl_xor(m, d, 32));
  int tile = b >> 4, bi = b & 15;
  f16* xrow = X + (((size_t)tile * T_ + t) * 16 + bi) * K_;
  v2h o;
  o[0] = (f16)expf(v0 - m);
  o[1] = (f16)expf(v1 - m);
  *(v2h*)(xrow + 2 * lane) = o;
  if (lane == 0) ME[((size_t)tile * T_ + t) * 16 + bi] = m;
}

// ---------------------------------------------------------------------------
// Kernel 2: numerator, one wave per batch; lanes stride over time steps so
// tag/mask loads are coalesced, then xor-shuffle reduction.
// ---------------------------------------------------------------------------
__global__ void crf_numer(const float* __restrict__ em, const int* __restrict__ tags,
                          const int* __restrict__ mask, const float* __restrict__ st,
                          const float* __restrict__ en, const float* __restrict__ tr,
                          float* __restrict__ num) {
  int w = threadIdx.x >> 5, lane = threadIdx.x & 31;
  int b = blockIdx.x * 8 + w;                // one wave per batch
  const float* e  = em   + (size_t)b * T_ * K_;
  const int*   tg = tags + (size_t)b * T_;
  const int*   mk = mask + (size_t)b * T_;
  float s = 0.0f;
  int cnt = 0;
  for (int t = lane; t < T_; t += 32) {
    int tt = tg[t];
    int m  = mk[t];
    if (t == 0) {
      s += st[tt] + e[tt];                   // start + emit[0], unmasked
    } else if (m) {
      int pv = tg[t - 1];
      s += tr[pv * K_ + tt] + e[(size_t)t * K_ + tt];
    }
    cnt += m;
  }
  for (int d = 1; d < 32; d <<= 1) {
    s   += __shfl_xor(s, d, 32);
    cnt += __shfl_xor(cnt, d, 32);
  }
  if (lane == 0) num[b] = s + en[tg[cnt - 1]];
}

// ---------------------------------------------------------------------------
// Kernel 3: forward scan via WMMA. One WG = 16-batch tile, 4 waves = 4
// 16-class column slices. alpha[b,:] = c_b + log U[b,:].
//   A (constant/wave) = Ehat^T chunk, f16 16x32
//   B (per step)      = U^T chunk from LDS, f16 32x16
//   D = V[class, batch] f32 16x16  -> W = V*X, r = max, U' = W/r, c += me+log r
// ---------------------------------------------------------------------------
__global__ void __launch_bounds__(128) crf_scan(
    const f16* __restrict__ X, const float* __restrict__ ME,
    const int* __restrict__ mask, const float* __restrict__ st,
    const float* __restrict__ en, const float* __restrict__ tr,
    float* __restrict__ den) {
  __shared__ __align__(16) f16 Ubuf[16 * UP];   // [batch][class], padded
  __shared__ float pmax[64];                     // [wave][batch]
  const int tid  = threadIdx.x;
  const int w    = tid >> 5, lane = tid & 31;
  const int h    = lane >> 4, bl = lane & 15;    // half-wave, batch-in-tile
  const int tile = blockIdx.x;
  const int j    = 16 * w + bl;                  // A-matrix row: output class

  // Constant A fragments: A0 covers alpha classes 0..31, A1 covers 32..63.
  // 16-bit A 16x32 layout (contiguous, matching B): lane<16: elems 0..15 -> K 0..15;
  // lane>=16: elems 0..15 -> K 16..31.
  v16h a0, a1;
  for (int e = 0; e < 16; ++e) {
    int kk = e + 16 * h;                         // alpha class within chunk
    a0[e] = (f16)expf(tr[kk * K_ + j]);
    a1[e] = (f16)expf(tr[(kk + 32) * K_ + j]);
  }

  const int cls = 16 * w + 8 * h;                // this lane's D classes cls..cls+7
  const f16* Xb = X + (((size_t)tile * T_) * 16 + bl) * K_;
  const int* mkp = mask + (tile * 16 + bl) * T_;

  // t = 0: U0 = exp(start) * X[0],  c = ME[0]
  {
    v8h xv = *(const v8h*)(Xb + cls);
    v8h u;
    for (int r = 0; r < 8; ++r) u[r] = (f16)(expf(st[cls + r]) * (float)xv[r]);
    *(v8h*)&Ubuf[bl * UP + cls] = u;
  }
  float c = 0.0f;
  if (w == 0 && h == 0) c = ME[((size_t)tile * T_) * 16 + bl];
  __syncthreads();

  for (int t = 1; t < T_; ++t) {
    // B fragments (U^T): B0 elem e <-> alpha class e+16h; B1 same +32.
    v8h u00 = *(const v8h*)&Ubuf[bl * UP + 16 * h];
    v8h u01 = *(const v8h*)&Ubuf[bl * UP + 16 * h + 8];
    v8h u10 = *(const v8h*)&Ubuf[bl * UP + 32 + 16 * h];
    v8h u11 = *(const v8h*)&Ubuf[bl * UP + 32 + 16 * h + 8];
    v16h b0, b1;
    for (int r = 0; r < 8; ++r) {
      b0[r] = u00[r]; b0[r + 8] = u01[r];
      b1[r] = u10[r]; b1[r + 8] = u11[r];
    }
    const f16* xp = Xb + (size_t)t * (16 * K_) + cls;
    v8h xv = *(const v8h*)xp;
    __builtin_prefetch(xp + 16 * K_, 0, 1);      // next step's X line
    int mk = mkp[t];

    v8f acc = {};
    acc = __builtin_amdgcn_wmma_f32_16x16x32_f16(false, a0, false, b0,
                                                 (short)0, acc, false, false);
    acc = __builtin_amdgcn_wmma_f32_16x16x32_f16(false, a1, false, b1,
                                                 (short)0, acc, false, false);

    float wv[8];
    float m = 0.0f;                              // values are nonnegative
    for (int r = 0; r < 8; ++r) { wv[r] = acc[r] * (float)xv[r]; m = fmaxf(m, wv[r]); }
    m = fmaxf(m, __shfl_xor(m, 16, 32));         // merge both class halves
    if (h == 0) pmax[w * 16 + bl] = m;
    __syncthreads();                             // pmax RAW + Ubuf WAR fence
    float rmx = fmaxf(fmaxf(pmax[bl], pmax[16 + bl]),
                      fmaxf(pmax[32 + bl], pmax[48 + bl]));
    // rcp instead of IEEE divide: rescale error is ~1 ulp/step and cancels in
    // the log-domain accumulation (we add log(rmx) for exactly this rmx).
    float inv = __builtin_amdgcn_rcpf(rmx);
    if (mk) {
      v8h u;
      for (int r = 0; r < 8; ++r) u[r] = (f16)(wv[r] * inv);
      *(v8h*)&Ubuf[bl * UP + cls] = u;
      if (w == 0 && h == 0)
        c += ME[((size_t)tile * T_ + t) * 16 + bl] + logf(rmx);
    }
    __syncthreads();                             // Ubuf RAW fence for next step
  }

  // denominator: den = c + log sum_j U[b][j] * exp(end[j])
  if (w == 0 && h == 0) {
    float s = 0.0f;
    for (int jj = 0; jj < K_; ++jj)
      s += (float)Ubuf[bl * UP + jj] * expf(en[jj]);
    den[tile * 16 + bl] = c + logf(s);
  }
}

// ---------------------------------------------------------------------------
// Kernel 4: mean over batches of (num - den).
// ---------------------------------------------------------------------------
__global__ void crf_final(const float* __restrict__ num, const float* __restrict__ den,
                          float* __restrict__ out) {
  __shared__ float red[256];
  float s = 0.0f;
  for (int i = threadIdx.x; i < B_; i += 256) s += num[i] - den[i];
  red[threadIdx.x] = s;
  __syncthreads();
  for (int off = 128; off > 0; off >>= 1) {
    if (threadIdx.x < off) red[threadIdx.x] += red[threadIdx.x + off];
    __syncthreads();
  }
  if (threadIdx.x == 0) out[0] = red[0] * (1.0f / (float)B_);
}

// ---------------------------------------------------------------------------
extern "C" void kernel_launch(void* const* d_in, const int* in_sizes, int n_in,
                              void* d_out, int out_size, void* d_ws, size_t ws_size,
                              hipStream_t stream) {
  const float* emissions = (const float*)d_in[0];
  const int*   tags      = (const int*)d_in[1];
  const int*   mask      = (const int*)d_in[2];
  const float* startt    = (const float*)d_in[3];
  const float* endt      = (const float*)d_in[4];
  const float* trans     = (const float*)d_in[5];

  char*  ws  = (char*)d_ws;
  f16*   X   = (f16*)ws;                          // 64 MB
  float* ME  = (float*)(ws + XBYTES);             // 2 MB
  float* num = (float*)(ws + XBYTES + MEBYTES);   // 4 KB
  float* den = num + B_;                          // 4 KB

  crf_prep <<<dim3((B_ * T_) / 8), dim3(256), 0, stream>>>(emissions, X, ME);
  crf_numer<<<dim3(B_ / 8),        dim3(256), 0, stream>>>(emissions, tags, mask,
                                                           startt, endt, trans, num);
  crf_scan <<<dim3(B_ / 16),       dim3(128), 0, stream>>>(X, ME, mask, startt,
                                                           endt, trans, den);
  crf_final<<<dim3(1),             dim3(256), 0, stream>>>(num, den, (float*)d_out);
}
